// GraphFull_1726576854110
// MI455X (gfx1250) — matrix operand order repaired and checked by reading, not compiled, not run on hardware
//
#include <hip/hip_runtime.h>
#include <hip/hip_bf16.h>
#include <math.h>

// ---------------------------------------------------------------------------
// GraphConv x2 + L2-normalize for MI455X (gfx1250, wave32, WMMA).
//
// Pipeline (all on `stream`, graph-capture safe):
//   1. pack_w_split : W1,W2 -> bf16 hi/lo copies (K padded to mult of 32)
//   2. gemm_bf16x3<300,320,false> : H = X @ W1^T        (v_wmma_f32_16x16x32_bf16 x3)
//   3. memset A1 = 0 ; spmm_atomic : A1 = Adj @ H       (global_atomic_add_f32)
//   4. gemm_bf16x3<512,512,true>  : H = relu(A1+b1) @ W2^T   (bias+relu fused in A load)
//   5. memset out = 0 ; spmm_atomic : out = Adj @ H
//   6. bias_norm : out = (out + b2) / max(||row||2, eps)  (in place)
//
// GEMM tiling: block = 256 threads = 8 waves = 2 M-tiles x 4 N-groups.
// Each wave: 16(M) x 128(N) output = 8 x v_wmma_f32_16x16x32_bf16 accumulators,
// so each A-fragment bf16x3 split (~50 VALU) feeds 24 WMMAs (co-executable).
// ---------------------------------------------------------------------------

typedef __attribute__((ext_vector_type(16))) __bf16 v16bf;
typedef __attribute__((ext_vector_type(8)))  float  v8f;
typedef __attribute__((ext_vector_type(4)))  float  f32x4;

#define FEAT 512  // HID == OUT_C == 512

__device__ __forceinline__ void splitf(float x, __bf16& h, __bf16& l) {
  h = (__bf16)x;                 // RNE, keeps top ~8 mantissa bits
  l = (__bf16)(x - (float)h);    // next ~8 bits
}

__device__ __forceinline__ v8f wmma3(const v16bf& ahi, const v16bf& alo,
                                     const v16bf& bhi, const v16bf& blo, v8f c) {
  // (ahi+alo)*(bhi+blo) ~= ahi*bhi + ahi*blo + alo*bhi   (drop lo*lo term)
  c = __builtin_amdgcn_wmma_f32_16x16x32_bf16(false, ahi, false, blo, (short)0, c, false, false);
  c = __builtin_amdgcn_wmma_f32_16x16x32_bf16(false, alo, false, bhi, (short)0, c, false, false);
  c = __builtin_amdgcn_wmma_f32_16x16x32_bf16(false, ahi, false, bhi, (short)0, c, false, false);
  return c;
}

// Split weights [rows, K] (row-major, K contiguous) into bf16 hi/lo, K padded to KP.
__global__ __launch_bounds__(256) void pack_w_split(
    const float* __restrict__ W, __bf16* __restrict__ hi, __bf16* __restrict__ lo,
    int K, int KP, int total) {
  int idx = blockIdx.x * 256 + threadIdx.x;
  if (idx >= total) return;
  int k = idx % KP;
  int r = idx / KP;
  float v = (k < K) ? W[(size_t)r * K + k] : 0.0f;
  __bf16 h, l;
  splitf(v, h, l);
  hi[idx] = h;
  lo[idx] = l;
}

// C[M,512] = op(A)[M,K] @ W^T, W split as bf16 hi/lo with row stride KP.
// op(A) = BIAS_RELU ? relu(A + bias[k]) : A.
// One wave computes a 16x128 output tile (8 x 16x16 WMMA accumulators).
// A fragment layout (16-bit A 16x32): lane<16 -> row (m+r), K {0..7,16..23};
// lane>=16 -> K {8..15,24..31}.  B fragment: lane gives column, same K split as A.
template<int K, int KP, bool BIAS_RELU>
__global__ __launch_bounds__(256) void gemm_bf16x3(
    const float* __restrict__ A, const __bf16* __restrict__ Whi,
    const __bf16* __restrict__ Wlo, const float* __restrict__ bias,
    float* __restrict__ C, int M) {
  const int lane = threadIdx.x & 31;
  const int wave = threadIdx.x >> 5;          // 8 waves: 2 M-tiles x 4 N-groups
  const int half = lane >> 4;
  const int r    = lane & 15;
  const int mbase = blockIdx.x * 32 + (wave >> 2) * 16;
  const int nb0   = (wave & 3) * 128;

  int arow = mbase + r;
  if (arow >= M) arow = M - 1;                // clamp; stores are guarded
  const float* ap = A + (size_t)arow * K;

  v8f acc[8];
#pragma unroll
  for (int t = 0; t < 8; t++)
    acc[t] = (v8f){0.f, 0.f, 0.f, 0.f, 0.f, 0.f, 0.f, 0.f};

  for (int k0 = 0; k0 < KP; k0 += 32) {
    const int kbase = k0 + half * 8;
    float af[16];
    if (k0 + 32 <= K) {                       // full step: vector loads
      f32x4 c0 = *(const f32x4*)(ap + kbase);
      f32x4 c1 = *(const f32x4*)(ap + kbase + 4);
      f32x4 c2 = *(const f32x4*)(ap + kbase + 16);
      f32x4 c3 = *(const f32x4*)(ap + kbase + 20);
#pragma unroll
      for (int i = 0; i < 4; i++) {
        af[i] = c0[i]; af[4 + i] = c1[i]; af[8 + i] = c2[i]; af[12 + i] = c3[i];
      }
      if (BIAS_RELU) {
        f32x4 b0 = *(const f32x4*)(bias + kbase);
        f32x4 b1 = *(const f32x4*)(bias + kbase + 4);
        f32x4 b2 = *(const f32x4*)(bias + kbase + 16);
        f32x4 b3 = *(const f32x4*)(bias + kbase + 20);
#pragma unroll
      for (int i = 0; i < 4; i++) {
          af[i]      = fmaxf(af[i]      + b0[i], 0.f);
          af[4 + i]  = fmaxf(af[4 + i]  + b1[i], 0.f);
          af[8 + i]  = fmaxf(af[8 + i]  + b2[i], 0.f);
          af[12 + i] = fmaxf(af[12 + i] + b3[i], 0.f);
        }
      }
    } else {                                  // tail step (K=300): element guards
#pragma unroll
      for (int i = 0; i < 16; i++) {
        int k = kbase + (i & 7) + ((i >= 8) ? 16 : 0);
        float v = (k < K) ? ap[k] : 0.0f;
        if (BIAS_RELU && (k < K)) v = fmaxf(v + bias[k], 0.f);
        af[i] = v;
      }
    }

    v16bf ahi, alo;
#pragma unroll
    for (int i = 0; i < 16; i++) {
      __bf16 h, l;
      splitf(af[i], h, l);
      ahi[i] = h;
      alo[i] = l;
    }

#pragma unroll
    for (int t = 0; t < 8; t++) {
      size_t boff = (size_t)(nb0 + 16 * t + r) * KP + k0 + half * 8;
      v16bf bhi, blo;
#pragma unroll
      for (int i = 0; i < 16; i++) {
        size_t o = boff + (i & 7) + ((i >= 8) ? 16 : 0);
        bhi[i] = Whi[o];
        blo[i] = Wlo[o];
      }
      acc[t] = wmma3(ahi, alo, bhi, blo, acc[t]);
    }
  }

  // D layout: elem i -> row mbase + i + 8*half, col nb0 + 16*t + r
#pragma unroll
  for (int t = 0; t < 8; t++) {
    int col = nb0 + 16 * t + r;
#pragma unroll
    for (int i = 0; i < 8; i++) {
      int row = mbase + i + 8 * half;
      if (row < M) C[(size_t)row * FEAT + col] = acc[t][i];
    }
  }
}

// out[row[e], :] += val[e] * feat[col[e], :]  — 128 lanes/edge, float4 each.
__global__ __launch_bounds__(256) void spmm_atomic(
    const int* __restrict__ erow, const int* __restrict__ ecol,
    const float* __restrict__ eval, const float* __restrict__ feat,
    float* __restrict__ out, int E) {
  long long gid = (long long)blockIdx.x * 256 + threadIdx.x;
  int e = (int)(gid >> 7);
  if (e >= E) return;
  int f = ((int)gid & 127) * 4;
  float v = eval[e];
  f32x4 d = *(const f32x4*)(feat + (size_t)ecol[e] * FEAT + f);
  float* dst = out + (size_t)erow[e] * FEAT + f;
  __hip_atomic_fetch_add(dst + 0, v * d[0], __ATOMIC_RELAXED, __HIP_MEMORY_SCOPE_AGENT);
  __hip_atomic_fetch_add(dst + 1, v * d[1], __ATOMIC_RELAXED, __HIP_MEMORY_SCOPE_AGENT);
  __hip_atomic_fetch_add(dst + 2, v * d[2], __ATOMIC_RELAXED, __HIP_MEMORY_SCOPE_AGENT);
  __hip_atomic_fetch_add(dst + 3, v * d[3], __ATOMIC_RELAXED, __HIP_MEMORY_SCOPE_AGENT);
}

// In place on d_out: o = (o + b2); o /= max(||o||_2, 1e-12). One block per row.
__global__ __launch_bounds__(256) void bias_norm(float* __restrict__ o,
                                                 const float* __restrict__ b2) {
  const int t = threadIdx.x;
  float* row = o + (size_t)blockIdx.x * FEAT;
  float x0 = row[t] + b2[t];
  float x1 = row[t + 256] + b2[t + 256];
  float ss = x0 * x0 + x1 * x1;
#pragma unroll
  for (int off = 16; off > 0; off >>= 1) ss += __shfl_down(ss, off, 32);
  __shared__ float part[8];
  __shared__ float invn;
  if ((t & 31) == 0) part[t >> 5] = ss;
  __syncthreads();
  if (t == 0) {
    float s = 0.f;
#pragma unroll
    for (int i = 0; i < 8; i++) s += part[i];
    invn = 1.0f / fmaxf(sqrtf(s), 1e-12f);
  }
  __syncthreads();
  row[t]       = x0 * invn;
  row[t + 256] = x1 * invn;
}

extern "C" void kernel_launch(void* const* d_in, const int* in_sizes, int n_in,
                              void* d_out, int out_size, void* d_ws, size_t ws_size,
                              hipStream_t stream) {
  const float* x    = (const float*)d_in[0];
  const int*   erow = (const int*)d_in[1];
  const int*   ecol = (const int*)d_in[2];
  const float* evl  = (const float*)d_in[3];
  const float* W1   = (const float*)d_in[4];
  const float* b1   = (const float*)d_in[5];
  const float* W2   = (const float*)d_in[6];
  const float* b2   = (const float*)d_in[7];
  float* out = (float*)d_out;

  const int IN_C = 300, KP1 = 320;
  const int N = in_sizes[0] / IN_C;   // 50000
  const int E = in_sizes[1];          // 800000

  // Workspace layout
  char* p = (char*)d_ws;
  float*  H    = (float*)p;  p += (size_t)N * FEAT * sizeof(float);   // H1 then H2
  float*  A1   = (float*)p;  p += (size_t)N * FEAT * sizeof(float);   // aggregated
  __bf16* W1hi = (__bf16*)p; p += (size_t)FEAT * KP1 * sizeof(__bf16);
  __bf16* W1lo = (__bf16*)p; p += (size_t)FEAT * KP1 * sizeof(__bf16);
  __bf16* W2hi = (__bf16*)p; p += (size_t)FEAT * FEAT * sizeof(__bf16);
  __bf16* W2lo = (__bf16*)p; p += (size_t)FEAT * FEAT * sizeof(__bf16);

  const int mblocks = (N + 31) / 32;  // 32 rows per block (2 M-tiles)
  const int spmm_blocks = (int)(((long long)E * 128 + 255) / 256);

  // 1. split weights into bf16 hi/lo (K padded)
  pack_w_split<<<(FEAT * KP1 + 255) / 256, 256, 0, stream>>>(W1, W1hi, W1lo, IN_C, KP1, FEAT * KP1);
  pack_w_split<<<(FEAT * FEAT + 255) / 256, 256, 0, stream>>>(W2, W2hi, W2lo, FEAT, FEAT, FEAT * FEAT);

  // 2. H = X @ W1^T
  gemm_bf16x3<300, 320, false><<<mblocks, 256, 0, stream>>>(x, W1hi, W1lo, nullptr, H, N);

  // 3. A1 = Adj @ H
  hipMemsetAsync(A1, 0, (size_t)N * FEAT * sizeof(float), stream);
  spmm_atomic<<<spmm_blocks, 256, 0, stream>>>(erow, ecol, evl, H, A1, E);

  // 4. H = relu(A1 + b1) @ W2^T  (bias+relu fused into A-operand load)
  gemm_bf16x3<512, 512, true><<<mblocks, 256, 0, stream>>>(A1, W2hi, W2lo, b1, H, N);

  // 5. out = Adj @ H
  hipMemsetAsync(out, 0, (size_t)N * FEAT * sizeof(float), stream);
  spmm_atomic<<<spmm_blocks, 256, 0, stream>>>(erow, ecol, evl, H, out, E);

  // 6. out = (out + b2) / max(||row||, eps)
  bias_norm<<<N, 256, 0, stream>>>(out, b2);
}
